// MixtralSparseMoeBlock_9371618640144
// MI455X (gfx1250) — matrix operand order repaired and checked by reading, not compiled, not run on hardware
//
#include <hip/hip_runtime.h>
#include <hip/hip_bf16.h>
#include <stdint.h>
#include <stddef.h>

#define HDIM 1024
#define FDIM 3584
#define NEXP 8
#define NTOK 2048
#define NROWS (NTOK * 2)   // top-2 => exactly 4096 expert-rows total
#define BP    40           // padded LDS pitch (ushorts) for B tiles

typedef __attribute__((ext_vector_type(16))) __bf16 v16bf;
typedef __attribute__((ext_vector_type(8)))  float  v8f;
typedef __attribute__((ext_vector_type(4)))  int    v4i;

union FragBF { uint4 u[2]; v16bf v; };

// Async global->LDS (CDNA5), confirmed working on this toolchain:
//   void __builtin_amdgcn_global_load_async_to_lds_b128(v4i as1*, v4i as3*, imm, imm)
#if defined(__HIP_DEVICE_COMPILE__) && \
    __has_builtin(__builtin_amdgcn_global_load_async_to_lds_b128) && \
    __has_builtin(__builtin_amdgcn_s_wait_asynccnt)
#define USE_ASYNC_LDS 1
#else
#define USE_ASYNC_LDS 0
#endif

__device__ __forceinline__ void g2l_b128(unsigned short* l, const unsigned short* g) {
#if USE_ASYNC_LDS
  __builtin_amdgcn_global_load_async_to_lds_b128(
      (__attribute__((address_space(1))) v4i*)g,
      (__attribute__((address_space(3))) v4i*)l, 0, 0);
#else
  *(uint4*)l = *(const uint4*)g;
#endif
}
__device__ __forceinline__ void async_fence() {
#if USE_ASYNC_LDS
  __builtin_amdgcn_s_wait_asynccnt(0);
#endif
}

// fp32 -> bf16, round-half-up at the 16-bit boundary.
// One v_add per element; the >>16 folds into a *_d16_hi store.
__device__ __forceinline__ unsigned short bf16_rn(float f) {
  union { float f; unsigned u; } c; c.f = f;
  return (unsigned short)((c.u + 0x8000u) >> 16);
}

// ---------------------------------------------------------------- init
__global__ void k_init(unsigned* cnt) {
  if (threadIdx.x < NEXP) cnt[threadIdx.x] = 0u;
}

// ---------------------------------------------------------------- router
__global__ __launch_bounds__(256) void k_router(
    const float* __restrict__ x, const float* __restrict__ gw,
    float* __restrict__ out, float* __restrict__ logits_out,
    unsigned short* __restrict__ xbf,
    int* __restrict__ sel, float* __restrict__ wts, int* __restrict__ pos,
    unsigned* __restrict__ cnt)
{
  const int wave = threadIdx.x >> 5, lane = threadIdx.x & 31;
  const int t = blockIdx.x * 8 + wave;
  const float* xt = x + (size_t)t * HDIM;

  float acc[NEXP];
#pragma unroll
  for (int e = 0; e < NEXP; ++e) acc[e] = 0.f;

  for (int h = lane; h < HDIM; h += 32) {
    float xv = xt[h];
    xbf[(size_t)t * HDIM + h] = bf16_rn(xv);
    out[(size_t)t * HDIM + h] = 0.f;                // zero-init accumulator
    const float* g = gw + (size_t)h * NEXP;
#pragma unroll
    for (int e = 0; e < NEXP; ++e) acc[e] += xv * g[e];
  }
#pragma unroll
  for (int off = 16; off > 0; off >>= 1)
#pragma unroll
    for (int e = 0; e < NEXP; ++e) acc[e] += __shfl_xor(acc[e], off, 32);

  if (lane == 0) {
    float mx = acc[0];
#pragma unroll
    for (int e = 1; e < NEXP; ++e) mx = fmaxf(mx, acc[e]);
    float p[NEXP], s = 0.f;
#pragma unroll
    for (int e = 0; e < NEXP; ++e) { p[e] = __expf(acc[e] - mx); s += p[e]; }
    float inv = 1.f / s;
#pragma unroll
    for (int e = 0; e < NEXP; ++e) {
      p[e] *= inv;
      logits_out[t * NEXP + e] = acc[e];            // raw logits (output #2)
    }
    int i0 = 0;
#pragma unroll
    for (int e = 1; e < NEXP; ++e) if (p[e] > p[i0]) i0 = e;
    int i1 = (i0 == 0) ? 1 : 0;
#pragma unroll
    for (int e = 0; e < NEXP; ++e) if (e != i0 && p[e] > p[i1]) i1 = e;
    float w0 = p[i0], w1 = p[i1], wn = 1.f / (w0 + w1);
    sel[t * 2 + 0] = i0; wts[t * 2 + 0] = w0 * wn;
    sel[t * 2 + 1] = i1; wts[t * 2 + 1] = w1 * wn;
    pos[t * 2 + 0] = (int)atomicAdd(&cnt[i0], 1u);
    pos[t * 2 + 1] = (int)atomicAdd(&cnt[i1], 1u);
  }
}

// ---------------------------------------------------------------- prefix + rows
__global__ void k_prefix(const unsigned* __restrict__ cnt, int* __restrict__ base) {
  if (threadIdx.x == 0) {
    int b = 0;
    for (int e = 0; e < NEXP; ++e) { base[e] = b; b += (int)cnt[e]; }
  }
}

__global__ void k_rows(const int* __restrict__ sel, const float* __restrict__ wts,
                       const int* __restrict__ pos, const int* __restrict__ base,
                       int* __restrict__ row_token, float* __restrict__ row_w)
{
  int t = blockIdx.x * 256 + threadIdx.x;
  if (t >= NTOK) return;
#pragma unroll
  for (int k = 0; k < 2; ++k) {
    int e = sel[t * 2 + k];
    int r = base[e] + pos[t * 2 + k];
    row_token[r] = t;
    row_w[r] = wts[t * 2 + k];
  }
}

// ---------------------------------------------------------------- phase A
__global__ __launch_bounds__(256) void k_upgate(
    const unsigned short* __restrict__ xbf,
    const float* __restrict__ up_w, const float* __restrict__ gp_w,
    const int* __restrict__ row_token,
    const unsigned* __restrict__ cnt, const int* __restrict__ base,
    unsigned short* __restrict__ inner)
{
  __shared__ unsigned short As[2][64 * 32];
  __shared__ unsigned short Bu[2][64 * BP];
  __shared__ unsigned short Bg[2][64 * BP];

  const int e  = blockIdx.z;
  const int rt = blockIdx.y;
  const int ft = blockIdx.x;
  const int cntE = (int)cnt[e];
  const int r0 = rt * 64;
  if (r0 >= cntE) return;
  const int baseE = base[e];
  const int f0 = ft * 64;

  const int tid = threadIdx.x;
  const int wave = tid >> 5, lane = tid & 31;
  const int mr = wave & 3;
  const int nc = wave >> 2;

  const int aOff0 = (mr * 16 + (lane & 15)) * 32 + (lane >> 4) * 16;
  const int aOff1 = aOff0 + 8;
  const int bOff0 = (nc * 32 + (lane & 15)) * BP + (lane >> 4) * 16;
  const int bOff1 = bOff0 + 16 * BP;

  const int sar = tid >> 2;
  const int sac = (tid & 3) * 8;
  int rg = r0 + sar; if (rg >= cntE) rg = 0;
  const unsigned short* aS = xbf + (size_t)row_token[baseE + rg] * HDIM + sac;
  const int aLds = sar * 32 + sac;

  const int bk = tid >> 4;
  const int bf = (tid & 15) * 4;
  const float* upE = up_w + (size_t)e * HDIM * FDIM;
  const float* gpE = gp_w + (size_t)e * HDIM * FDIM;
  const float* pu = upE + (size_t)bk * FDIM + (f0 + bf);
  const float* pg = gpE + (size_t)bk * FDIM + (f0 + bf);

  auto stage = [&](int b, const unsigned short* aPtr, const float* u, const float* g) {
    g2l_b128(&As[b][aLds], aPtr);
#pragma unroll
    for (int i = 0; i < 2; ++i) {
      float4 u4 = *(const float4*)(u + (size_t)i * 16 * FDIM);
      float4 g4 = *(const float4*)(g + (size_t)i * 16 * FDIM);
      int k = bk + i * 16;
      unsigned short* bub = &Bu[b][bf * BP + k];
      unsigned short* bgb = &Bg[b][bf * BP + k];
      bub[0 * BP] = bf16_rn(u4.x);
      bub[1 * BP] = bf16_rn(u4.y);
      bub[2 * BP] = bf16_rn(u4.z);
      bub[3 * BP] = bf16_rn(u4.w);
      bgb[0 * BP] = bf16_rn(g4.x);
      bgb[1 * BP] = bf16_rn(g4.y);
      bgb[2 * BP] = bf16_rn(g4.z);
      bgb[3 * BP] = bf16_rn(g4.w);
    }
  };

  v8f accU0 = {}, accU1 = {}, accG0 = {}, accG1 = {};

  constexpr int NIT = HDIM / 32;
  stage(0, aS, pu, pg);
  aS += 32; pu += (size_t)32 * FDIM; pg += (size_t)32 * FDIM;
  int buf = 0;

  for (int it = 0; it < NIT; ++it) {
    async_fence();
    __syncthreads();
    if (it + 1 < NIT) {
      stage(buf ^ 1, aS, pu, pg);
      if (it + 2 < NIT && (tid & 3) == 0) {
        __builtin_prefetch(pu + (size_t)32 * FDIM, 0, 1);
        __builtin_prefetch(pg + (size_t)32 * FDIM, 0, 1);
      }
      aS += 32; pu += (size_t)32 * FDIM; pg += (size_t)32 * FDIM;
    }
    const unsigned short* a = As[buf];
    const unsigned short* bu = Bu[buf];
    const unsigned short* bg = Bg[buf];
    FragBF fa, fu0, fu1, fg0, fg1;
    fa.u[0]  = *(const uint4*)(a + aOff0);   fa.u[1]  = *(const uint4*)(a + aOff1);
    fu0.u[0] = *(const uint4*)(bu + bOff0);  fu0.u[1] = *(const uint4*)(bu + bOff0 + 8);
    fu1.u[0] = *(const uint4*)(bu + bOff1);  fu1.u[1] = *(const uint4*)(bu + bOff1 + 8);
    fg0.u[0] = *(const uint4*)(bg + bOff0);  fg0.u[1] = *(const uint4*)(bg + bOff0 + 8);
    fg1.u[0] = *(const uint4*)(bg + bOff1);  fg1.u[1] = *(const uint4*)(bg + bOff1 + 8);
    accU0 = __builtin_amdgcn_wmma_f32_16x16x32_bf16(false, fa.v, false, fu0.v, (short)0, accU0, false, false);
    accU1 = __builtin_amdgcn_wmma_f32_16x16x32_bf16(false, fa.v, false, fu1.v, (short)0, accU1, false, false);
    accG0 = __builtin_amdgcn_wmma_f32_16x16x32_bf16(false, fa.v, false, fg0.v, (short)0, accG0, false, false);
    accG1 = __builtin_amdgcn_wmma_f32_16x16x32_bf16(false, fa.v, false, fg1.v, (short)0, accG1, false, false);
    buf ^= 1;
  }

  const int cn  = lane & 15;
  const int mhi = (lane >> 4) * 8;
  const int col = f0 + nc * 32 + cn;
#pragma unroll
  for (int i = 0; i < 8; ++i) {
    int r = r0 + mr * 16 + mhi + i;
    if (r < cntE) {
      float u0 = accU0[i], u1 = accU1[i];
      float s0 = (u0 / (1.f + __expf(-u0))) * accG0[i];
      float s1 = (u1 / (1.f + __expf(-u1))) * accG1[i];
      unsigned short* p = inner + (size_t)(baseE + r) * FDIM + col;
      p[0]  = bf16_rn(s0);
      p[16] = bf16_rn(s1);
    }
  }
}

// ---------------------------------------------------------------- phase B
__global__ __launch_bounds__(256) void k_down(
    const unsigned short* __restrict__ inner,
    const float* __restrict__ down_w,
    const int* __restrict__ row_token, const float* __restrict__ row_w,
    const unsigned* __restrict__ cnt, const int* __restrict__ base,
    float* __restrict__ out)
{
  __shared__ unsigned short As[2][64 * 32];
  __shared__ unsigned short Bd[2][64 * BP];

  const int e  = blockIdx.z;
  const int rt = blockIdx.y;
  const int ht = blockIdx.x;
  const int cntE = (int)cnt[e];
  const int r0 = rt * 64;
  if (r0 >= cntE) return;
  const int baseE = base[e];
  const int h0 = ht * 64;

  const int tid = threadIdx.x;
  const int wave = tid >> 5, lane = tid & 31;
  const int mr = wave & 3, nc = wave >> 2;

  const int aOff0 = (mr * 16 + (lane & 15)) * 32 + (lane >> 4) * 16;
  const int aOff1 = aOff0 + 8;
  const int bOff0 = (nc * 32 + (lane & 15)) * BP + (lane >> 4) * 16;
  const int bOff1 = bOff0 + 16 * BP;

  const int sar = tid >> 2;
  const int sac = (tid & 3) * 8;
  int rr = r0 + sar; if (rr >= cntE) rr = cntE - 1;
  const unsigned short* aS = inner + (size_t)(baseE + rr) * FDIM + sac;
  const int aLds = sar * 32 + sac;

  const int bk = tid >> 4;
  const int bf = (tid & 15) * 4;
  const float* dwE = down_w + (size_t)e * FDIM * HDIM;
  const float* pd = dwE + (size_t)bk * HDIM + (h0 + bf);

  auto stage = [&](int b, const unsigned short* aPtr, const float* d) {
    g2l_b128(&As[b][aLds], aPtr);
#pragma unroll
    for (int i = 0; i < 2; ++i) {
      float4 d4 = *(const float4*)(d + (size_t)i * 16 * HDIM);
      int k = bk + i * 16;
      unsigned short* bdb = &Bd[b][bf * BP + k];
      bdb[0 * BP] = bf16_rn(d4.x);
      bdb[1 * BP] = bf16_rn(d4.y);
      bdb[2 * BP] = bf16_rn(d4.z);
      bdb[3 * BP] = bf16_rn(d4.w);
    }
  };

  v8f accD0 = {}, accD1 = {};

  constexpr int NIT = FDIM / 32;
  stage(0, aS, pd);
  aS += 32; pd += (size_t)32 * HDIM;
  int buf = 0;

  for (int it = 0; it < NIT; ++it) {
    async_fence();
    __syncthreads();
    if (it + 1 < NIT) {
      stage(buf ^ 1, aS, pd);
      if (it + 2 < NIT && (tid & 3) == 0)
        __builtin_prefetch(pd + (size_t)32 * HDIM, 0, 1);
      aS += 32; pd += (size_t)32 * HDIM;
    }
    const unsigned short* a = As[buf];
    const unsigned short* bd = Bd[buf];
    FragBF fa, fd0, fd1;
    fa.u[0]  = *(const uint4*)(a + aOff0);   fa.u[1]  = *(const uint4*)(a + aOff1);
    fd0.u[0] = *(const uint4*)(bd + bOff0);  fd0.u[1] = *(const uint4*)(bd + bOff0 + 8);
    fd1.u[0] = *(const uint4*)(bd + bOff1);  fd1.u[1] = *(const uint4*)(bd + bOff1 + 8);
    accD0 = __builtin_amdgcn_wmma_f32_16x16x32_bf16(false, fa.v, false, fd0.v, (short)0, accD0, false, false);
    accD1 = __builtin_amdgcn_wmma_f32_16x16x32_bf16(false, fa.v, false, fd1.v, (short)0, accD1, false, false);
    buf ^= 1;
  }

  const int cn  = lane & 15;
  const int mhi = (lane >> 4) * 8;
  const int col = h0 + nc * 32 + cn;
#pragma unroll
  for (int i = 0; i < 8; ++i) {
    int r = r0 + mr * 16 + mhi + i;
    if (r < cntE) {
      int   t = row_token[baseE + r];
      float w = row_w[baseE + r];
      float* o = out + (size_t)t * HDIM + col;
      atomicAdd(o,      accD0[i] * w);
      atomicAdd(o + 16, accD1[i] * w);
    }
  }
}

// ---------------------------------------------------------------- launch
extern "C" void kernel_launch(void* const* d_in, const int* in_sizes, int n_in,
                              void* d_out, int out_size, void* d_ws, size_t ws_size,
                              hipStream_t stream)
{
  (void)in_sizes; (void)n_in; (void)out_size; (void)ws_size;
  const float* x   = (const float*)d_in[0];
  const float* gw  = (const float*)d_in[1];
  const float* upw = (const float*)d_in[2];
  const float* gpw = (const float*)d_in[3];
  const float* dww = (const float*)d_in[4];

  float* out    = (float*)d_out;
  float* logits = out + (size_t)NTOK * HDIM;

  char* ws = (char*)d_ws;
  size_t o_xbf   = 0;
  size_t o_inner = o_xbf   + (size_t)NTOK * HDIM * 2;
  size_t o_rtok  = o_inner + (size_t)NROWS * FDIM * 2;
  size_t o_rw    = o_rtok  + (size_t)NROWS * 4;
  size_t o_sel   = o_rw    + (size_t)NROWS * 4;
  size_t o_wts   = o_sel   + (size_t)NTOK * 2 * 4;
  size_t o_pos   = o_wts   + (size_t)NTOK * 2 * 4;
  size_t o_cnt   = o_pos   + (size_t)NTOK * 2 * 4;
  size_t o_base  = o_cnt   + (size_t)NEXP * 4;

  unsigned short* xbf   = (unsigned short*)(ws + o_xbf);
  unsigned short* inner = (unsigned short*)(ws + o_inner);
  int*      row_token = (int*)(ws + o_rtok);
  float*    row_w     = (float*)(ws + o_rw);
  int*      sel       = (int*)(ws + o_sel);
  float*    wts       = (float*)(ws + o_wts);
  int*      pos       = (int*)(ws + o_pos);
  unsigned* cnt       = (unsigned*)(ws + o_cnt);
  int*      base      = (int*)(ws + o_base);

  k_init  <<<1, 32, 0, stream>>>(cnt);
  k_router<<<NTOK / 8, 256, 0, stream>>>(x, gw, out, logits, xbf, sel, wts, pos, cnt);
  k_prefix<<<1, 1, 0, stream>>>(cnt, base);
  k_rows  <<<(NTOK + 255) / 256, 256, 0, stream>>>(sel, wts, pos, base, row_token, row_w);

  dim3 gA(FDIM / 64, NTOK / 64, NEXP);
  k_upgate<<<gA, 256, 0, stream>>>(xbf, upw, gpw, row_token, cnt, base, inner);

  dim3 gB(HDIM / 64, NTOK / 64, NEXP);
  k_down<<<gB, 256, 0, stream>>>(inner, dww, row_token, row_w, cnt, base, out);
}
